// UnstackedGlm4vMoeTextMoE_2894807958211
// MI455X (gfx1250) — compile-verified, matrix operand order checked by reading, not run on hardware
//
#include <hip/hip_runtime.h>
#include <stdint.h>

// ---- problem constants (match reference config) ----
#define Tt 4096      // B*S tokens
#define Hh 1024      // hidden
#define Ii 768       // intermediate
#define Ee 32        // routed experts
#define Gg 8         // groups
#define MB 32        // tokens per GEMM workgroup (2 M-tiles of 16)
#define WAVES 8

typedef __attribute__((ext_vector_type(16))) __bf16 v16bf;
typedef __attribute__((ext_vector_type(8)))  float  v8f;

union FragAB { uint4 q[2]; v16bf v; };

__device__ __forceinline__ uint32_t f2bf_bits(float f) {
  uint32_t u = __builtin_bit_cast(uint32_t, f);
  u += 0x7FFFu + ((u >> 16) & 1u);          // round-to-nearest-even
  return u >> 16;
}

__device__ __forceinline__ __bf16 f2bf(float f) {
  uint16_t h = (uint16_t)f2bf_bits(f);
  return __builtin_bit_cast(__bf16, h);
}

// ---------------- fp32 -> bf16 conversion (one HBM pass, 4 elems/thread-iter) ----
__global__ void k_cvt4(const float4* __restrict__ src, uint2* __restrict__ dst, size_t n4) {
  size_t i = (size_t)blockIdx.x * blockDim.x + threadIdx.x;
  size_t stride = (size_t)gridDim.x * blockDim.x;
  for (; i < n4; i += stride) {
    float4 a = src[i];
    uint2 o;
    o.x = (f2bf_bits(a.y) << 16) | f2bf_bits(a.x);
    o.y = (f2bf_bits(a.w) << 16) | f2bf_bits(a.z);
    dst[i] = o;
  }
}

// ---------------- init per-expert counters ----------------
__global__ void k_init(int* cnt) {
  int i = threadIdx.x;
  if (i < Ee) cnt[i] = 0;
  if (i == Ee) cnt[Ee] = Tt;   // "expert 32" = shared expert over all tokens
}

// ---------------- router: one wave32 per token ----------------
__global__ __launch_bounds__(32) void k_router(const float* __restrict__ X,
    const float* __restrict__ RW, const float* __restrict__ bias,
    int* __restrict__ cnt, int* __restrict__ etok, float* __restrict__ ew) {
  int t = blockIdx.x;
  int e = threadIdx.x;                 // lane = expert
  const float4* xp = (const float4*)(X + (size_t)t * Hh);
  const float4* wp = (const float4*)(RW + (size_t)e * Hh);
  float acc = 0.f;
  #pragma unroll 4
  for (int k = 0; k < Hh / 4; ++k) {
    float4 a = xp[k], b = wp[k];
    acc += a.x * b.x + a.y * b.y + a.z * b.z + a.w * b.w;
  }
  float s = 1.0f / (1.0f + __expf(-acc));   // keep exact div: feeds routing compares
  __shared__ float ss[Ee], ssc[Ee];
  ss[e]  = s;
  ssc[e] = s + bias[e];
  __syncthreads();
  if (e == 0) {
    // group scores = sum of top-2 corrected scores within each group of 4
    float gs[Gg];
    for (int g = 0; g < Gg; ++g) {
      float v[4];
      for (int j = 0; j < 4; ++j) v[j] = ssc[g * 4 + j];
      int i1 = 0;
      for (int j = 1; j < 4; ++j) if (v[j] > v[i1]) i1 = j;
      float m1 = v[i1]; v[i1] = -1e30f;
      int i2 = 0;
      for (int j = 1; j < 4; ++j) if (v[j] > v[i2]) i2 = j;
      gs[g] = m1 + v[i2];
    }
    // top-4 groups (strict argmax == jax tie-break on lower index)
    bool gsel[Gg];
    for (int g = 0; g < Gg; ++g) gsel[g] = false;
    for (int r = 0; r < 4; ++r) {
      int bi = 0; float bv = -1e30f;
      for (int g = 0; g < Gg; ++g) if (!gsel[g] && gs[g] > bv) { bv = gs[g]; bi = g; }
      gsel[bi] = true;
    }
    float scc[Ee];
    for (int i = 0; i < Ee; ++i) scc[i] = gsel[i >> 2] ? ssc[i] : 0.0f;
    int idx[4]; float w[4]; float wsum = 1e-20f;
    for (int r = 0; r < 4; ++r) {
      int bi = 0; float bv = -1e30f;
      for (int i = 0; i < Ee; ++i) if (scc[i] > bv) { bv = scc[i]; bi = i; }
      scc[bi] = -1e30f;
      idx[r] = bi; w[r] = ss[bi]; wsum += ss[bi];
    }
    for (int r = 0; r < 4; ++r) {
      float wr = w[r] / wsum;                       // ROUTED_SCALE = 1.0
      int pos = atomicAdd(&cnt[idx[r]], 1);
      etok[idx[r] * Tt + pos] = (r << 16) | t;      // pack slot|token
      ew  [idx[r] * Tt + pos] = wr;
    }
    etok[Ee * Tt + t] = (4 << 16) | t;              // shared-expert slot 4
    ew  [Ee * Tt + t] = 1.0f;
  }
}

// ---------------- grouped expert SwiGLU MLP, WMMA bf16 + async LDS staging ----
__global__ __launch_bounds__(256) void k_expert(
    const __bf16* __restrict__ Xb,
    const __bf16* __restrict__ Wg, const __bf16* __restrict__ Wu, const __bf16* __restrict__ Wd,
    const __bf16* __restrict__ SWg, const __bf16* __restrict__ SWu, const __bf16* __restrict__ SWd,
    const int* __restrict__ cnt, const int* __restrict__ etok, const float* __restrict__ ew,
    float* __restrict__ yslots) {
  int e = blockIdx.y;
  int n = cnt[e];
  int base = blockIdx.x * MB;
  if (base >= n) return;

  const __bf16* wg = (e < Ee) ? Wg + (size_t)e * Ii * Hh : SWg;
  const __bf16* wu = (e < Ee) ? Wu + (size_t)e * Ii * Hh : SWu;
  const __bf16* wd = (e < Ee) ? Wd + (size_t)e * Hh * Ii : SWd;

  __shared__ __bf16 sX[MB * Hh];       // 64 KB: gathered activation tile (bf16)
  __shared__ __bf16 sh[MB * Ii];       // 48 KB: bf16 intermediate h
  __shared__ int   s_tok[MB];
  __shared__ int   s_slot[MB];
  __shared__ float s_w[MB];
  __shared__ int   s_valid[MB];

  int tid = threadIdx.x;
  if (tid < MB) {
    int p = base + tid;
    int q = (p < n) ? p : (n - 1);     // pad rows duplicate last valid token
    int packed = etok[e * Tt + q];
    s_tok[tid]   = packed & 0xFFFF;
    s_slot[tid]  = packed >> 16;
    s_w[tid]     = ew[e * Tt + q];
    s_valid[tid] = (p < n) ? 1 : 0;
  }
  __syncthreads();

  // ---- async-stage the gathered X rows into LDS (CDNA5 ASYNCcnt path) ----
  {
    unsigned ldsBase = (unsigned)(uintptr_t)&sX[0];   // low 32 bits = LDS offset
    const char* xbase = (const char*)Xb;
    #pragma unroll
    for (int it = 0; it < (MB * Hh * 2) / (256 * 16); ++it) {   // 16 x 16B per thread
      int f = tid + it * 256;            // 16-byte chunk index within tile
      int row = f >> 7;                  // 128 chunks per 2048-byte row
      int cb  = (f & 127) << 4;          // byte offset within row
      unsigned lds = ldsBase + ((unsigned)f << 4);
      unsigned long long ga =
          (unsigned long long)(uintptr_t)(xbase + (size_t)s_tok[row] * (Hh * 2) + cb);
      asm volatile("global_load_async_to_lds_b128 %0, %1, off"
                   :: "v"(lds), "v"(ga) : "memory");
    }
    asm volatile("s_wait_asynccnt 0x0" ::: "memory");
  }
  __syncthreads();

  int wave = tid >> 5;
  int lane = tid & 31;
  int l    = lane & 15;
  int hi   = lane >> 4;                // lane group: K-halves split per ISA layout
  int koff = hi ? 8 : 0;

  // ---- phase 1: h = silu(x Wg^T) * (x Wu^T); 2 M-tiles x 48 N-tiles ----
  for (int tile = wave; tile < 2 * (Ii / 16); tile += WAVES) {
    int mt = tile & 1;
    int nt = tile >> 1;
    v8f accg = {}; v8f accu = {};
    const __bf16* pa = &sX[(size_t)(mt * 16 + l) * Hh + koff];      // A from LDS
    const __bf16* pg = wg + (size_t)(nt * 16 + l) * Hh + koff;
    const __bf16* pu = wu + (size_t)(nt * 16 + l) * Hh + koff;
    if (tile + WAVES < 2 * (Ii / 16)) {                             // prefetch next B tiles
      int nt2 = (tile + WAVES) >> 1;
      __builtin_prefetch(wg + (size_t)(nt2 * 16 + l) * Hh, 0, 1);
      __builtin_prefetch(wu + (size_t)(nt2 * 16 + l) * Hh, 0, 1);
    }
    #pragma unroll 4
    for (int kk = 0; kk < Hh / 32; ++kk) {
      FragAB fa, fg, fu;
      fa.q[0] = *(const uint4*)(pa); fa.q[1] = *(const uint4*)(pa + 16);
      fg.q[0] = *(const uint4*)(pg); fg.q[1] = *(const uint4*)(pg + 16);
      fu.q[0] = *(const uint4*)(pu); fu.q[1] = *(const uint4*)(pu + 16);
      accg = __builtin_amdgcn_wmma_f32_16x16x32_bf16(false, fa.v, false, fg.v,
                                                     (short)0, accg, false, false);
      accu = __builtin_amdgcn_wmma_f32_16x16x32_bf16(false, fa.v, false, fu.v,
                                                     (short)0, accu, false, false);
      pa += 32; pg += 32; pu += 32;
    }
    #pragma unroll
    for (int i = 0; i < 8; ++i) {
      int m = i + (hi ? 8 : 0);        // C/D layout: VGPR i -> M=i (+8 for hi lanes)
      float g = accg[i], u = accu[i];
      // silu(g)*u with HW v_rcp (result is bf16-quantized anyway)
      float hval = g * __builtin_amdgcn_rcpf(1.0f + __expf(-g)) * u;
      sh[(mt * 16 + m) * Ii + nt * 16 + l] = f2bf(hval);
    }
  }
  __syncthreads();

  // ---- phase 2: y = (h Wd^T) * combine_w; 2 M-tiles x 64 N-tiles ----
  for (int tile = wave; tile < 2 * (Hh / 16); tile += WAVES) {
    int mt = tile & 1;
    int nt = tile >> 1;
    v8f acc = {};
    const __bf16* pl = &sh[(size_t)(mt * 16 + l) * Ii + koff];      // A from LDS
    const __bf16* pb = wd + (size_t)(nt * 16 + l) * Ii + koff;
    if (tile + WAVES < 2 * (Hh / 16)) {
      int nt2 = (tile + WAVES) >> 1;
      __builtin_prefetch(wd + (size_t)(nt2 * 16 + l) * Ii, 0, 1);
    }
    #pragma unroll 4
    for (int kk = 0; kk < Ii / 32; ++kk) {
      FragAB fa, fb;
      fa.q[0] = *(const uint4*)(pl); fa.q[1] = *(const uint4*)(pl + 16);
      fb.q[0] = *(const uint4*)(pb); fb.q[1] = *(const uint4*)(pb + 16);
      acc = __builtin_amdgcn_wmma_f32_16x16x32_bf16(false, fa.v, false, fb.v,
                                                    (short)0, acc, false, false);
      pl += 32; pb += 32;
    }
    #pragma unroll
    for (int i = 0; i < 8; ++i) {
      int m = i + (hi ? 8 : 0);
      int r = mt * 16 + m;
      if (s_valid[r]) {
        int tok = s_tok[r], slot = s_slot[r];
        yslots[((size_t)slot * Tt + tok) * Hh + nt * 16 + l] = s_w[r] * acc[i];
      }
    }
  }
}

// ---------------- final slot reduction: out = sum of 5 slots ----------------
__global__ void k_reduce(const float* __restrict__ ys, float* __restrict__ out) {
  size_t n = (size_t)Tt * Hh;
  size_t stride = (size_t)gridDim.x * blockDim.x;
  for (size_t i = (size_t)blockIdx.x * blockDim.x + threadIdx.x; i < n; i += stride) {
    out[i] = ys[i] + ys[n + i] + ys[2 * n + i] + ys[3 * n + i] + ys[4 * n + i];
  }
}

extern "C" void kernel_launch(void* const* d_in, const int* in_sizes, int n_in,
                              void* d_out, int out_size, void* d_ws, size_t ws_size,
                              hipStream_t stream) {
  (void)in_sizes; (void)n_in; (void)out_size; (void)ws_size;
  const float* X   = (const float*)d_in[0];   // hidden_states [B,S,H]
  const float* RW  = (const float*)d_in[1];   // router_weight [E,H]
  const float* BIA = (const float*)d_in[2];   // e_score_correction_bias [E]
  const float* GW  = (const float*)d_in[3];   // gate_w [E,I,H]
  const float* UW  = (const float*)d_in[4];   // up_w [E,I,H]
  const float* DW  = (const float*)d_in[5];   // down_w [E,H,I]
  const float* SGW = (const float*)d_in[6];   // shared_gate_w [I,H]
  const float* SUW = (const float*)d_in[7];   // shared_up_w [I,H]
  const float* SDW = (const float*)d_in[8];   // shared_down_w [H,I]
  float* out = (float*)d_out;

  char* ws = (char*)d_ws;
  size_t off = 0;
  auto take = [&](size_t bytes) -> char* {
    char* p = ws + off;
    off = (off + bytes + 255) & ~(size_t)255;
    return p;
  };
  __bf16* Xb  = (__bf16*)take((size_t)Tt * Hh * 2);
  __bf16* Wg  = (__bf16*)take((size_t)Ee * Ii * Hh * 2);
  __bf16* Wu  = (__bf16*)take((size_t)Ee * Ii * Hh * 2);
  __bf16* Wd  = (__bf16*)take((size_t)Ee * Hh * Ii * 2);
  __bf16* SWg = (__bf16*)take((size_t)Ii * Hh * 2);
  __bf16* SWu = (__bf16*)take((size_t)Ii * Hh * 2);
  __bf16* SWd = (__bf16*)take((size_t)Hh * Ii * 2);
  int*    cnt  = (int*)  take((size_t)(Ee + 1) * 4);
  int*    etok = (int*)  take((size_t)(Ee + 1) * Tt * 4);
  float*  ew   = (float*)take((size_t)(Ee + 1) * Tt * 4);
  float*  yslots = (float*)take((size_t)5 * Tt * Hh * 4);

  // 1) one-pass fp32 -> bf16 (result set fits in 192MB L2 for GEMM re-reads)
  k_cvt4<<<1024, 256, 0, stream>>>((const float4*)X,   (uint2*)Xb,  (size_t)Tt * Hh / 4);
  k_cvt4<<<4096, 256, 0, stream>>>((const float4*)GW,  (uint2*)Wg,  (size_t)Ee * Ii * Hh / 4);
  k_cvt4<<<4096, 256, 0, stream>>>((const float4*)UW,  (uint2*)Wu,  (size_t)Ee * Ii * Hh / 4);
  k_cvt4<<<4096, 256, 0, stream>>>((const float4*)DW,  (uint2*)Wd,  (size_t)Ee * Hh * Ii / 4);
  k_cvt4<<<256,  256, 0, stream>>>((const float4*)SGW, (uint2*)SWg, (size_t)Ii * Hh / 4);
  k_cvt4<<<256,  256, 0, stream>>>((const float4*)SUW, (uint2*)SWu, (size_t)Ii * Hh / 4);
  k_cvt4<<<256,  256, 0, stream>>>((const float4*)SDW, (uint2*)SWd, (size_t)Hh * Ii / 4);

  // 2) routing + per-expert token gather
  k_init<<<1, 64, 0, stream>>>(cnt);
  k_router<<<Tt, 32, 0, stream>>>(X, RW, BIA, cnt, etok, ew);

  // 3) grouped SwiGLU expert GEMMs (shared expert = expert 32), WMMA bf16
  dim3 ggrid(Tt / MB, Ee + 1);
  k_expert<<<ggrid, 256, 0, stream>>>(Xb, Wg, Wu, Wd, SWg, SWu, SWd,
                                      cnt, etok, ew, yslots);

  // 4) combine 4 routed slots + shared slot
  k_reduce<<<4096, 256, 0, stream>>>(yslots, out);
}
